// EfficientFormerMeta3D_59614146068631
// MI455X (gfx1250) — matrix operand order repaired and checked by reading, not compiled, not run on hardware
//
#include <hip/hip_runtime.h>
#include <math.h>

// ---------------- types ----------------
typedef __bf16 bf16;
typedef bf16  v16bf __attribute__((ext_vector_type(16)));
typedef bf16  v8bf  __attribute__((ext_vector_type(8)));
typedef float v8f   __attribute__((ext_vector_type(8)));

#define B_   256
#define S_   49
#define D_   448
#define H_   8
#define KD_  32
#define ED_  128
#define QKV_OUT_ (H_ * (KD_ + KD_ + ED_))   // 1536
#define MLP_HID_ (4 * D_)                   // 1792
#define ROWS_ (B_ * S_)                     // 12544
#define SCALE_ 0.17677669529663687f         // 32^-0.5
#define EPS_ 1e-12f

// ---------------- helpers ----------------
__device__ __forceinline__ float gelu_exact(float x) {
    return 0.5f * x * (1.0f + erff(x * 0.70710678118654752f));
}

// ---------------- f32 -> bf16 convert ----------------
__global__ void k_cvt_bf16(const float* __restrict__ in, bf16* __restrict__ out, int n) {
    int i = blockIdx.x * blockDim.x + threadIdx.x;
    if (i < n) out[i] = (bf16)in[i];
}

// ---------------- LayerNorm (1 wave per row, D=448 = 14*32) ----------------
__global__ void k_layernorm_bf16(const float* __restrict__ x,
                                 const float* __restrict__ w,
                                 const float* __restrict__ b,
                                 bf16* __restrict__ out) {
    int row  = blockIdx.x * (blockDim.x >> 5) + (threadIdx.x >> 5);
    int lane = threadIdx.x & 31;
    if (row >= ROWS_) return;
    const float* xr = x + (size_t)row * D_;
    float v[14];
    float s = 0.0f;
#pragma unroll
    for (int i = 0; i < 14; ++i) { v[i] = xr[lane + 32 * i]; s += v[i]; }
#pragma unroll
    for (int off = 16; off > 0; off >>= 1) s += __shfl_xor(s, off, 32);
    float mu = s * (1.0f / (float)D_);
    float q = 0.0f;
#pragma unroll
    for (int i = 0; i < 14; ++i) { float d = v[i] - mu; q += d * d; }
#pragma unroll
    for (int off = 16; off > 0; off >>= 1) q += __shfl_xor(q, off, 32);
    float inv = rsqrtf(q * (1.0f / (float)D_) + EPS_);
    bf16* orow = out + (size_t)row * D_;
#pragma unroll
    for (int i = 0; i < 14; ++i) {
        int c = lane + 32 * i;
        orow[c] = (bf16)((v[i] - mu) * inv * w[c] + b[c]);
    }
}

// ---------------- bf16 WMMA GEMM: C[M,N] = A[M,K] * W[N,K]^T + bias ----------------
// One wave computes a 16(M) x 64(N) tile with 4 accumulators (A fragment reused 4x).
// EPI: 1 = store bf16
//      2 = GELU then store bf16
//      3 = out_f32[m,n] = resid[m,n] + scale[n] * (acc + bias)   (proj + residual)
//      4 = out_f32[m,n] += scale[n] * (acc + bias)               (fc2 + residual acc)
template <int EPI>
__global__ void k_gemm_bf16_wmma(const bf16* __restrict__ A,
                                 const bf16* __restrict__ W,
                                 const float* __restrict__ bias,
                                 void* __restrict__ Cout,
                                 const float* __restrict__ resid,
                                 const float* __restrict__ scale,
                                 int M, int N, int K, int totalWaves) {
    int wave = blockIdx.x * (blockDim.x >> 5) + (threadIdx.x >> 5);
    if (wave >= totalWaves) return;
    int lane = threadIdx.x & 31;
    int nGroups = N >> 6;
    int mTile = wave / nGroups;
    int nGroup = wave - mTile * nGroups;
    int m0 = mTile << 4;
    int n0 = nGroup << 6;
    int kh   = lane >> 4;   // which K-half this lane holds
    int lmod = lane & 15;

    // A fragment base: row m0+lmod, 16 contiguous K at kh*16 (matches B layout)
    const bf16* Arow  = A + (size_t)(m0 + lmod) * K + kh * 16;
    // B fragment: lane holds 16 contiguous K at kh*16 of row n (same contiguous k order as A)
    const bf16* Wbase = W + kh * 16;

    v8f acc[4];
#pragma unroll
    for (int t = 0; t < 4; ++t) acc[t] = (v8f){0,0,0,0,0,0,0,0};

    for (int k = 0; k < K; k += 32) {
        v8bf a0 = *(const v8bf*)(Arow + k);
        v8bf a1 = *(const v8bf*)(Arow + k + 8);
        v16bf a;
#pragma unroll
        for (int e = 0; e < 8; ++e) { a[e] = a0[e]; a[e + 8] = a1[e]; }
        v16bf b0 = *(const v16bf*)(Wbase + (size_t)(n0 +  0 + lmod) * K + k);
        v16bf b1 = *(const v16bf*)(Wbase + (size_t)(n0 + 16 + lmod) * K + k);
        v16bf b2 = *(const v16bf*)(Wbase + (size_t)(n0 + 32 + lmod) * K + k);
        v16bf b3 = *(const v16bf*)(Wbase + (size_t)(n0 + 48 + lmod) * K + k);
        acc[0] = __builtin_amdgcn_wmma_f32_16x16x32_bf16(false, a, false, b0, (short)0, acc[0], false, false);
        acc[1] = __builtin_amdgcn_wmma_f32_16x16x32_bf16(false, a, false, b1, (short)0, acc[1], false, false);
        acc[2] = __builtin_amdgcn_wmma_f32_16x16x32_bf16(false, a, false, b2, (short)0, acc[2], false, false);
        acc[3] = __builtin_amdgcn_wmma_f32_16x16x32_bf16(false, a, false, b3, (short)0, acc[3], false, false);
    }

    // Epilogue. C/D layout: lane n = lmod, VGPR r holds row m = m0 + kh*8 + r.
    int mbase = m0 + kh * 8;
#pragma unroll
    for (int nt = 0; nt < 4; ++nt) {
        int n = n0 + nt * 16 + lmod;
        float bv = bias ? bias[n] : 0.0f;
        float sv = (EPI >= 3) ? scale[n] : 0.0f;
#pragma unroll
        for (int r = 0; r < 8; ++r) {
            float val = acc[nt][r] + bv;
            size_t idx = (size_t)(mbase + r) * N + n;
            if (EPI == 1) {
                ((bf16*)Cout)[idx] = (bf16)val;
            } else if (EPI == 2) {
                ((bf16*)Cout)[idx] = (bf16)gelu_exact(val);
            } else if (EPI == 3) {
                ((float*)Cout)[idx] = resid[idx] + sv * val;
            } else { // EPI == 4
                ((float*)Cout)[idx] += sv * val;
            }
        }
    }
}

// ---------------- attention: one block per (b,h) ----------------
// qkv row layout: [ROWS_, 1536], per head h: q at h*192, k at h*192+32, v at h*192+64.
// ctx output: bf16 [ROWS_, H_*ED_] (row = b*49+s, col = h*128+e)
__global__ void k_attention(const bf16* __restrict__ qkv,
                            const float* __restrict__ biases,   // [H, 49]
                            const int* __restrict__ idxs,       // [49*49]
                            bf16* __restrict__ ctx) {
    __shared__ float sq[S_][KD_];
    __shared__ float sk[S_][KD_];
    __shared__ float sv[S_][ED_];
    __shared__ float sc[S_][S_ + 3];
    __shared__ float brow[S_];

    int b = blockIdx.x >> 3;
    int h = blockIdx.x & 7;
    int tid = threadIdx.x;
    size_t rowbase = (size_t)b * S_;

    // stage q/k/v for this (b,h) into LDS (49*192 = 9408 elems)
    for (int i = tid; i < S_ * 192; i += blockDim.x) {
        int s = i / 192;
        int d = i - s * 192;
        float val = (float)qkv[(rowbase + s) * QKV_OUT_ + (size_t)h * 192 + d];
        if (d < KD_)            sq[s][d] = val;
        else if (d < 2 * KD_)   sk[s][d - KD_] = val;
        else                    sv[s][d - 2 * KD_] = val;
    }
    for (int i = tid; i < S_; i += blockDim.x) brow[i] = biases[h * S_ + i];
    __syncthreads();

    // scores = q @ k^T * scale + bias
    for (int i = tid; i < S_ * S_; i += blockDim.x) {
        int qi = i / S_;
        int kj = i - qi * S_;
        float acc = 0.0f;
#pragma unroll
        for (int d = 0; d < KD_; ++d) acc += sq[qi][d] * sk[kj][d];
        sc[qi][kj] = acc * SCALE_ + brow[idxs[i]];
    }
    __syncthreads();

    // row softmax (49 rows, one thread each)
    if (tid < S_) {
        float mx = -1e30f;
        for (int j = 0; j < S_; ++j) mx = fmaxf(mx, sc[tid][j]);
        float sum = 0.0f;
        for (int j = 0; j < S_; ++j) { float e = __expf(sc[tid][j] - mx); sc[tid][j] = e; sum += e; }
        float inv = 1.0f / sum;
        for (int j = 0; j < S_; ++j) sc[tid][j] *= inv;
    }
    __syncthreads();

    // ctx = probs @ v
    for (int i = tid; i < S_ * ED_; i += blockDim.x) {
        int qi = i / ED_;
        int e  = i - qi * ED_;
        float acc = 0.0f;
#pragma unroll 7
        for (int j = 0; j < S_; ++j) acc += sc[qi][j] * sv[j][e];
        ctx[(rowbase + qi) * (H_ * ED_) + (size_t)h * ED_ + e] = (bf16)acc;
    }
}

// ---------------- host launcher ----------------
static inline size_t align256(size_t x) { return (x + 255) & ~(size_t)255; }

extern "C" void kernel_launch(void* const* d_in, const int* in_sizes, int n_in,
                              void* d_out, int out_size, void* d_ws, size_t ws_size,
                              hipStream_t stream) {
    const float* hidden   = (const float*)d_in[0];
    const float* ln2_w    = (const float*)d_in[1];
    const float* ln2_b    = (const float*)d_in[2];
    const float* qkv_w    = (const float*)d_in[3];
    const float* qkv_b    = (const float*)d_in[4];
    const float* proj_w   = (const float*)d_in[5];
    const float* proj_b   = (const float*)d_in[6];
    const float* att_bias = (const float*)d_in[7];
    const float* fc1_w    = (const float*)d_in[8];
    const float* fc1_b    = (const float*)d_in[9];
    const float* fc2_w    = (const float*)d_in[10];
    const float* fc2_b    = (const float*)d_in[11];
    const float* ls1      = (const float*)d_in[12];
    const float* ls2      = (const float*)d_in[13];
    const int*   bidx     = (const int*)d_in[14];
    float* out = (float*)d_out;

    // workspace carve-up
    char* ws = (char*)d_ws;
    size_t off = 0;
    bf16* wQKV = (bf16*)(ws + off); off = align256(off + (size_t)QKV_OUT_ * D_ * 2);
    bf16* wPRJ = (bf16*)(ws + off); off = align256(off + (size_t)D_ * (H_ * ED_) * 2);
    bf16* wFC1 = (bf16*)(ws + off); off = align256(off + (size_t)MLP_HID_ * D_ * 2);
    bf16* wFC2 = (bf16*)(ws + off); off = align256(off + (size_t)D_ * MLP_HID_ * 2);
    bf16* normb = (bf16*)(ws + off); off = align256(off + (size_t)ROWS_ * D_ * 2);
    bf16* qkvo  = (bf16*)(ws + off); off = align256(off + (size_t)ROWS_ * QKV_OUT_ * 2);
    bf16* ctxb  = (bf16*)(ws + off); off = align256(off + (size_t)ROWS_ * (H_ * ED_) * 2);
    bf16* hmid  = (bf16*)(ws + off); off = align256(off + (size_t)ROWS_ * MLP_HID_ * 2);
    (void)ws_size;

    // 1) convert weights to bf16
    {
        int n;
        n = QKV_OUT_ * D_;      k_cvt_bf16<<<(n + 255) / 256, 256, 0, stream>>>(qkv_w, wQKV, n);
        n = D_ * (H_ * ED_);    k_cvt_bf16<<<(n + 255) / 256, 256, 0, stream>>>(proj_w, wPRJ, n);
        n = MLP_HID_ * D_;      k_cvt_bf16<<<(n + 255) / 256, 256, 0, stream>>>(fc1_w, wFC1, n);
        n = D_ * MLP_HID_;      k_cvt_bf16<<<(n + 255) / 256, 256, 0, stream>>>(fc2_w, wFC2, n);
    }

    // 2) layernorm -> bf16 (8 rows per 256-thread block)
    k_layernorm_bf16<<<(ROWS_ + 7) / 8, 256, 0, stream>>>(hidden, ln2_w, ln2_b, normb);

    // 3) qkv = norm @ qkv_w^T + b   -> bf16
    {
        int M = ROWS_, N = QKV_OUT_, K = D_;
        int waves = (M >> 4) * (N >> 6);
        k_gemm_bf16_wmma<1><<<(waves + 7) / 8, 256, 0, stream>>>(
            normb, wQKV, qkv_b, qkvo, nullptr, nullptr, M, N, K, waves);
    }

    // 4) attention -> ctx bf16
    k_attention<<<B_ * H_, 256, 0, stream>>>(qkvo, att_bias, bidx, ctxb);

    // 5) out = hidden + ls1 * (ctx @ proj_w^T + b)
    {
        int M = ROWS_, N = D_, K = H_ * ED_;
        int waves = (M >> 4) * (N >> 6);
        k_gemm_bf16_wmma<3><<<(waves + 7) / 8, 256, 0, stream>>>(
            ctxb, wPRJ, proj_b, out, hidden, ls1, M, N, K, waves);
    }

    // 6) hmid = gelu(norm @ fc1_w^T + b) -> bf16
    {
        int M = ROWS_, N = MLP_HID_, K = D_;
        int waves = (M >> 4) * (N >> 6);
        k_gemm_bf16_wmma<2><<<(waves + 7) / 8, 256, 0, stream>>>(
            normb, wFC1, fc1_b, hmid, nullptr, nullptr, M, N, K, waves);
    }

    // 7) out += ls2 * (hmid @ fc2_w^T + b)
    {
        int M = ROWS_, N = D_, K = MLP_HID_;
        int waves = (M >> 4) * (N >> 6);
        k_gemm_bf16_wmma<4><<<(waves + 7) / 8, 256, 0, stream>>>(
            hmid, wFC2, fc2_b, out, nullptr, ls2, M, N, K, waves);
    }
}
